// GCN_56203942036132
// MI455X (gfx1250) — matrix operand order repaired and checked by reading, not compiled, not run on hardware
//
#include <hip/hip_runtime.h>
#include <hip/hip_bf16.h>

#define NN 100000
#define NE 3200000
#define NG 256
#define NF 256
#define NC 16

typedef __attribute__((ext_vector_type(16))) __bf16 v16bf;
typedef __attribute__((ext_vector_type(8)))  float  v8f;

union BFrag { v16bf v; unsigned int u[8]; };

__device__ __forceinline__ unsigned int f2bf_pk(float a, float b) {
    // round-to-nearest-even f32 -> bf16, packed into one dword
    unsigned int ua = __float_as_uint(a);
    unsigned int ub = __float_as_uint(b);
    ua = (ua + 0x7fffu + ((ua >> 16) & 1u)) >> 16;
    ub = (ub + 0x7fffu + ((ub >> 16) & 1u)) >> 16;
    return (ua & 0xffffu) | (ub << 16);
}

// ---------------------------------------------------------------- utilities
__global__ __launch_bounds__(256) void gcn_zero_f4(float4* __restrict__ p, int n4) {
    int i = blockIdx.x * 256 + threadIdx.x;
    if (i < n4) p[i] = make_float4(0.f, 0.f, 0.f, 0.f);
}

__global__ __launch_bounds__(256) void gcn_degree(const int* __restrict__ esrc,
                                                  const int* __restrict__ edst,
                                                  int* __restrict__ dout,
                                                  int* __restrict__ din) {
    int e = blockIdx.x * 256 + threadIdx.x;
    if (e < NE) {
        atomicAdd(&dout[esrc[e]], 1);
        atomicAdd(&din[edst[e]], 1);
    }
}

__global__ __launch_bounds__(256) void gcn_norms(const int* __restrict__ dout,
                                                 const int* __restrict__ din,
                                                 float* __restrict__ nsrc,
                                                 float* __restrict__ ndst) {
    int i = blockIdx.x * 256 + threadIdx.x;
    if (i < NN) {
        int a = dout[i], b = din[i];
        nsrc[i] = (a > 0) ? rsqrtf((float)a) : 0.f;
        ndst[i] = (b > 0) ? rsqrtf((float)b) : 0.f;
    }
}

// single-block exclusive scan of in-degrees -> row_ptr (+ cursor copy)
__global__ __launch_bounds__(1024)
void gcn_scan(const int* __restrict__ cnt, int* __restrict__ rowp,
              int* __restrict__ cursor) {
    __shared__ int ssum[1024];
    const int CH = (NN + 1023) / 1024;            // 98
    int t = threadIdx.x;
    int lo = t * CH;
    int hi = (lo + CH < NN) ? (lo + CH) : NN;
    int s = 0;
    for (int i = lo; i < hi; ++i) s += cnt[i];
    ssum[t] = s;
    __syncthreads();
    for (int off = 1; off < 1024; off <<= 1) {    // Hillis-Steele inclusive scan
        int v = (t >= off) ? ssum[t - off] : 0;
        __syncthreads();
        ssum[t] += v;
        __syncthreads();
    }
    int run = (t == 0) ? 0 : ssum[t - 1];
    for (int i = lo; i < hi; ++i) {
        rowp[i] = run; cursor[i] = run;
        run += cnt[i];
    }
    if (t == 1023) rowp[NN] = run;                // == NE
}

__global__ __launch_bounds__(256) void gcn_fill(const int* __restrict__ esrc,
                                                const int* __restrict__ edst,
                                                int* __restrict__ cursor,
                                                int* __restrict__ cols) {
    int e = blockIdx.x * 256 + threadIdx.x;
    if (e < NE) {
        int pos = atomicAdd(&cursor[edst[e]], 1);
        cols[pos] = esrc[e];
    }
}

// W[k, n] (f32, k-major) -> Wt[n, k] (bf16, n-major) so B-fragment K-pairs are contiguous
__global__ __launch_bounds__(256) void gcn_cvt_w(const float* __restrict__ W,
                                                 __bf16* __restrict__ Wt) {
    int t = blockIdx.x * 256 + threadIdx.x;   // t = n*256 + k
    int n = t >> 8, k = t & 255;
    float f = W[k * NF + n];
    unsigned int u = __float_as_uint(f);
    u = (u + 0x7fffu + ((u >> 16) & 1u)) >> 16;
    unsigned short s = (unsigned short)u;
    Wt[t] = *reinterpret_cast<__bf16*>(&s);
}

// activations f32 -> bf16 (row-major layout preserved), 8 elements/thread
__global__ __launch_bounds__(256)
void gcn_cvt_x(const float* __restrict__ X, unsigned int* __restrict__ Xb) {
    int i = blockIdx.x * 256 + threadIdx.x;       // indexes groups of 8 floats
    if (i >= NN * NF / 8) return;
    const float* p = X + (size_t)i * 8;
    float4 t0 = *reinterpret_cast<const float4*>(p);
    float4 t1 = *reinterpret_cast<const float4*>(p + 4);
    uint4 o;
    o.x = f2bf_pk(t0.x, t0.y); o.y = f2bf_pk(t0.z, t0.w);
    o.z = f2bf_pk(t1.x, t1.y); o.w = f2bf_pk(t1.z, t1.w);
    *reinterpret_cast<uint4*>(Xb + (size_t)i * 4) = o;
}

// ---------------------------------------------------------------- WMMA GEMM
// Hn[row, n] = (X[row, :] @ W[:, n]) * nsrc[row]   (X pre-converted to bf16)
// grid.x = NN/16 row strips, block = 128 (4 waves), each wave owns 4 n-tiles.
__global__ __launch_bounds__(128)
void gcn_gemm_hn(const __bf16* __restrict__ Xb, const __bf16* __restrict__ Wt,
                 const float* __restrict__ nsrc, float* __restrict__ Hn) {
    const int lane = threadIdx.x & 31;
    const int wave = threadIdx.x >> 5;
    const int row0 = blockIdx.x * 16;
    const int m  = lane & 15;
    const int hf = lane >> 4;
    const int n0 = wave * 64 + m;            // this lane's column in tiles n0, +16, +32, +48

    const __bf16* xr = Xb + (size_t)(row0 + m) * NF;
    v8f acc0 = {}, acc1 = {}, acc2 = {}, acc3 = {};

    for (int k0 = 0; k0 < NF; k0 += 32) {
        BFrag a;
#pragma unroll
        for (int j = 0; j < 8; ++j) {
            // A (16-bit 16x32): K = 16*half + 2*j ; pairs contiguous
            int K = k0 + (hf << 4) + (j << 1);
            a.u[j] = *reinterpret_cast<const unsigned int*>(xr + K);
        }
        BFrag b0, b1, b2, b3;
#pragma unroll
        for (int j = 0; j < 8; ++j) {
            // B (32x16): lanes 0-15 -> K 0..15 over VGPRs, lanes 16-31 -> K 16..31
            int K = k0 + (hf << 4) + (j << 1);
            b0.u[j] = *reinterpret_cast<const unsigned int*>(Wt + (size_t)(n0 +  0) * NF + K);
            b1.u[j] = *reinterpret_cast<const unsigned int*>(Wt + (size_t)(n0 + 16) * NF + K);
            b2.u[j] = *reinterpret_cast<const unsigned int*>(Wt + (size_t)(n0 + 32) * NF + K);
            b3.u[j] = *reinterpret_cast<const unsigned int*>(Wt + (size_t)(n0 + 48) * NF + K);
        }
        acc0 = __builtin_amdgcn_wmma_f32_16x16x32_bf16(false, a.v, false, b0.v, (short)0, acc0, false, false);
        acc1 = __builtin_amdgcn_wmma_f32_16x16x32_bf16(false, a.v, false, b1.v, (short)0, acc1, false, false);
        acc2 = __builtin_amdgcn_wmma_f32_16x16x32_bf16(false, a.v, false, b2.v, (short)0, acc2, false, false);
        acc3 = __builtin_amdgcn_wmma_f32_16x16x32_bf16(false, a.v, false, b3.v, (short)0, acc3, false, false);
    }

    float ns[8];
#pragma unroll
    for (int j = 0; j < 8; ++j) ns[j] = nsrc[row0 + hf * 8 + j];

    float* out = Hn + (size_t)row0 * NF;
#pragma unroll
    for (int j = 0; j < 8; ++j) {
        // C/D layout: VGPR j holds row M = j + 8*half, col N = lane&15 (per tile)
        size_t base = (size_t)(hf * 8 + j) * NF;
        out[base + n0 +  0] = acc0[j] * ns[j];
        out[base + n0 + 16] = acc1[j] * ns[j];
        out[base + n0 + 32] = acc2[j] * ns[j];
        out[base + n0 + 48] = acc3[j] * ns[j];
    }
}

// ---------------------------------------------------------------- CSR aggregation
// one wave per destination node: gather hn rows of in-neighbors, accumulate in
// registers, write once with fused relu(acc * ndst + bias). No atomics.
__global__ __launch_bounds__(256)
void gcn_aggregate(const float* __restrict__ Hn, const int* __restrict__ rowp,
                   const int* __restrict__ cols, const float* __restrict__ ndst,
                   const float* __restrict__ bias, float* __restrict__ Out) {
    int lane = threadIdx.x & 31;
    int d = blockIdx.x * 8 + (threadIdx.x >> 5);
    if (d >= NN) return;
    int beg = rowp[d], end = rowp[d + 1];
    float a0 = 0.f, a1 = 0.f, a2 = 0.f, a3 = 0.f;
    float a4 = 0.f, a5 = 0.f, a6 = 0.f, a7 = 0.f;
    for (int i = beg; i < end; ++i) {
        int s = cols[i];
        const float* sp = Hn + (size_t)s * NF + lane * 8;
        float4 t0 = *reinterpret_cast<const float4*>(sp);
        float4 t1 = *reinterpret_cast<const float4*>(sp + 4);
        a0 += t0.x; a1 += t0.y; a2 += t0.z; a3 += t0.w;
        a4 += t1.x; a5 += t1.y; a6 += t1.z; a7 += t1.w;
    }
    float nd = ndst[d];
    const float* bp = bias + lane * 8;
    float4 b0 = *reinterpret_cast<const float4*>(bp);
    float4 b1 = *reinterpret_cast<const float4*>(bp + 4);
    float4 r0, r1;
    r0.x = fmaxf(a0 * nd + b0.x, 0.f); r0.y = fmaxf(a1 * nd + b0.y, 0.f);
    r0.z = fmaxf(a2 * nd + b0.z, 0.f); r0.w = fmaxf(a3 * nd + b0.w, 0.f);
    r1.x = fmaxf(a4 * nd + b1.x, 0.f); r1.y = fmaxf(a5 * nd + b1.y, 0.f);
    r1.z = fmaxf(a6 * nd + b1.z, 0.f); r1.w = fmaxf(a7 * nd + b1.w, 0.f);
    float* op = Out + (size_t)d * NF + lane * 8;
    *reinterpret_cast<float4*>(op)     = r0;
    *reinterpret_cast<float4*>(op + 4) = r1;
}

// ---------------------------------------------------------------- pooling + head
__global__ __launch_bounds__(256)
void gcn_pool_sum(const float* __restrict__ H, const int* __restrict__ gids,
                  float* __restrict__ sums) {
    int lane = threadIdx.x & 31;
    int node = blockIdx.x * 8 + (threadIdx.x >> 5);
    if (node >= NN) return;
    int g = gids[node];
    const float* sp = H + (size_t)node * NF + lane * 8;
    float4 t0 = *reinterpret_cast<const float4*>(sp);
    float4 t1 = *reinterpret_cast<const float4*>(sp + 4);
    float* dp = sums + (size_t)g * NF + lane * 8;
    unsafeAtomicAdd(dp + 0, t0.x); unsafeAtomicAdd(dp + 1, t0.y);
    unsafeAtomicAdd(dp + 2, t0.z); unsafeAtomicAdd(dp + 3, t0.w);
    unsafeAtomicAdd(dp + 4, t1.x); unsafeAtomicAdd(dp + 5, t1.y);
    unsafeAtomicAdd(dp + 6, t1.z); unsafeAtomicAdd(dp + 7, t1.w);
}

__global__ __launch_bounds__(256)
void gcn_pool_cnt(const int* __restrict__ gids, float* __restrict__ cnts) {
    int i = blockIdx.x * 256 + threadIdx.x;
    if (i < NN) unsafeAtomicAdd(&cnts[gids[i]], 1.0f);
}

// one block (32 lanes) per graph: 256x16 matvec + log_softmax over 16 lanes
__global__ __launch_bounds__(32)
void gcn_head(const float* __restrict__ sums, const float* __restrict__ cnts,
              const float* __restrict__ Wl, const float* __restrict__ bl,
              float* __restrict__ out) {
    int g = blockIdx.x;
    int lane = threadIdx.x;
    int k = lane & 15;
    float inv = 1.0f / fmaxf(cnts[g], 1.0f);
    const float* sg = sums + (size_t)g * NF;
    float acc = bl[k];
    for (int c = 0; c < NF; ++c)
        acc += sg[c] * inv * Wl[c * NC + k];
    float mx = acc;
#pragma unroll
    for (int off = 1; off < 16; off <<= 1) mx = fmaxf(mx, __shfl_xor(mx, off, 16));
    float ex = __expf(acc - mx);
    float sum = ex;
#pragma unroll
    for (int off = 1; off < 16; off <<= 1) sum += __shfl_xor(sum, off, 16);
    if (lane < 16) out[g * NC + k] = acc - mx - __logf(sum);
}

// ---------------------------------------------------------------- launch
extern "C" void kernel_launch(void* const* d_in, const int* in_sizes, int n_in,
                              void* d_out, int out_size, void* d_ws, size_t ws_size,
                              hipStream_t stream) {
    const float* feats = (const float*)d_in[0];
    const int*   esrc  = (const int*)d_in[1];
    const int*   edst  = (const int*)d_in[2];
    const int*   gids  = (const int*)d_in[3];
    const float* W1 = (const float*)d_in[4];  const float* b1 = (const float*)d_in[5];
    const float* W2 = (const float*)d_in[6];  const float* b2 = (const float*)d_in[7];
    const float* W3 = (const float*)d_in[8];  const float* b3 = (const float*)d_in[9];
    const float* Wl = (const float*)d_in[10]; const float* bl = (const float*)d_in[11];

    char* ws = (char*)d_ws;
    float*  bufA = (float*)ws;                                  // NN*NF f32 (hn)
    float*  bufB = (float*)(ws + (size_t)NN * NF * 4);          // NN*NF f32 (h)
    float*  nsrc = (float*)(ws + (size_t)2 * NN * NF * 4);      // NN
    float*  ndst = nsrc + NN;                                   // NN
    __bf16* Wt1  = (__bf16*)(ndst + NN);                        // NF*NF bf16 each
    __bf16* Wt2  = Wt1 + NF * NF;
    __bf16* Wt3  = Wt2 + NF * NF;
    float*  sums = (float*)(Wt3 + NF * NF);                     // NG*NF
    float*  cnts = sums + NG * NF;                              // NG
    int*    dout = (int*)(cnts + NG);                           // NN (out-degree)
    int*    din  = dout + NN;                                   // NN (in-degree)
    int*    rowp = din + NN;                                    // NN+1
    int*    curs = rowp + NN + 1;                               // NN
    int*    cols = curs + NN;                                   // NE
    __bf16* Xb   = (__bf16*)(cols + NE);                        // NN*NF bf16 activations

    const int nx8 = NN * NF / 8;      // 3,200,000 groups of 8 elements

    // CSR + norms (graph is fixed: build once, reuse for all 3 layers)
    gcn_zero_f4<<<dim3(2 * NN / 4 / 256 + 1), dim3(256), 0, stream>>>((float4*)dout, 2 * NN / 4);
    gcn_degree<<<dim3(NE / 256), dim3(256), 0, stream>>>(esrc, edst, dout, din);
    gcn_norms<<<dim3((NN + 255) / 256), dim3(256), 0, stream>>>(dout, din, nsrc, ndst);
    gcn_scan<<<dim3(1), dim3(1024), 0, stream>>>(din, rowp, curs);
    gcn_fill<<<dim3(NE / 256), dim3(256), 0, stream>>>(esrc, edst, curs, cols);

    // bf16 transposed weights
    gcn_cvt_w<<<dim3(NF * NF / 256), dim3(256), 0, stream>>>(W1, Wt1);
    gcn_cvt_w<<<dim3(NF * NF / 256), dim3(256), 0, stream>>>(W2, Wt2);
    gcn_cvt_w<<<dim3(NF * NF / 256), dim3(256), 0, stream>>>(W3, Wt3);

    // layer 1: feats -> bf16 -> WMMA GEMM (x norm_src) -> CSR aggregate+relu -> bufB
    gcn_cvt_x<<<dim3(nx8 / 256), dim3(256), 0, stream>>>(feats, (unsigned int*)Xb);
    gcn_gemm_hn<<<dim3(NN / 16), dim3(128), 0, stream>>>(Xb, Wt1, nsrc, bufA);
    gcn_aggregate<<<dim3(NN / 8), dim3(256), 0, stream>>>(bufA, rowp, cols, ndst, b1, bufB);

    // layer 2
    gcn_cvt_x<<<dim3(nx8 / 256), dim3(256), 0, stream>>>(bufB, (unsigned int*)Xb);
    gcn_gemm_hn<<<dim3(NN / 16), dim3(128), 0, stream>>>(Xb, Wt2, nsrc, bufA);
    gcn_aggregate<<<dim3(NN / 8), dim3(256), 0, stream>>>(bufA, rowp, cols, ndst, b2, bufB);

    // layer 3
    gcn_cvt_x<<<dim3(nx8 / 256), dim3(256), 0, stream>>>(bufB, (unsigned int*)Xb);
    gcn_gemm_hn<<<dim3(NN / 16), dim3(128), 0, stream>>>(Xb, Wt3, nsrc, bufA);
    gcn_aggregate<<<dim3(NN / 8), dim3(256), 0, stream>>>(bufA, rowp, cols, ndst, b3, bufB);

    // pooling + head
    gcn_zero_f4<<<dim3(((NG * NF + NG) / 4 + 255) / 256), dim3(256), 0, stream>>>((float4*)sums, (NG * NF + NG) / 4);
    gcn_pool_sum<<<dim3(NN / 8), dim3(256), 0, stream>>>(bufB, gids, sums);
    gcn_pool_cnt<<<dim3((NN + 255) / 256), dim3(256), 0, stream>>>(gids, cnts);
    gcn_head<<<dim3(NG), dim3(32), 0, stream>>>(sums, cnts, Wl, bl, (float*)d_out);
}
